// CascadePredictionModel_76450417868972
// MI455X (gfx1250) — matrix operand order repaired and checked by reading, not compiled, not run on hardware
//
#include <hip/hip_runtime.h>

#define NN 10000   // nodes
#define NE 160000  // edges
#define CC 512     // channels (in = out)
#define MM 2000    // missing nodes
#define KK 512     // GEMM K

typedef __bf16 bf16_t;
typedef __attribute__((ext_vector_type(16))) __bf16 v16bf;
typedef __attribute__((ext_vector_type(8)))  __bf16 v8bf;
typedef __attribute__((ext_vector_type(8)))  float  v8f;

union V16U { v16bf v; v8bf h[2]; };

// ---------------------------------------------------------------- graph norm
__global__ __launch_bounds__(256) void k_deg_init(float* deg) {
    int i = blockIdx.x * 256 + threadIdx.x;
    if (i < NN) deg[i] = 1.0f;  // self loop contributes 1 to every node
}

__global__ __launch_bounds__(256) void k_deg_accum(const int* __restrict__ dst, float* deg) {
    int e = blockIdx.x * 256 + threadIdx.x;
    if (e < NE) atomicAdd(&deg[dst[e]], 1.0f);
}

__global__ __launch_bounds__(256) void k_dis(float* deg) {
    int i = blockIdx.x * 256 + threadIdx.x;
    if (i < NN) deg[i] = rsqrtf(deg[i]);  // deg >= 1 always
}

__global__ __launch_bounds__(256) void k_norm(const int* __restrict__ src,
                                              const int* __restrict__ dst,
                                              const float* __restrict__ dis,
                                              float* __restrict__ nrm) {
    int e = blockIdx.x * 256 + threadIdx.x;
    if (e < NE) nrm[e] = dis[src[e]] * dis[dst[e]];
}

// ------------------------------------------------------------- conversions
__global__ __launch_bounds__(256) void k_f32_to_bf16(const float* __restrict__ in,
                                                     bf16_t* __restrict__ out, int n) {
    int i = blockIdx.x * 256 + threadIdx.x;
    if (i < n) out[i] = (bf16_t)in[i];
}

// W[k][n] (KK x CC, row-major) -> Wt[n][k] bf16 (CC x KK, row-major)
__global__ __launch_bounds__(256) void k_w_transpose(const float* __restrict__ W,
                                                     bf16_t* __restrict__ Wt) {
    int idx = blockIdx.x * 256 + threadIdx.x;
    if (idx < KK * CC) {
        int n = idx & (CC - 1);
        int k = idx >> 9;
        Wt[(size_t)n * KK + k] = (bf16_t)W[(size_t)k * CC + n];
    }
}

__global__ __launch_bounds__(256) void k_relu_to_bf16(const float* __restrict__ in,
                                                      bf16_t* __restrict__ out, int n) {
    int i = blockIdx.x * 256 + threadIdx.x;
    if (i < n) out[i] = (bf16_t)fmaxf(in[i], 0.0f);
}

__global__ __launch_bounds__(256) void k_relu_inplace(float* p, int n) {
    int i = blockIdx.x * 256 + threadIdx.x;
    if (i < n) p[i] = fmaxf(p[i], 0.0f);
}

// ------------------------------------------------------------------- GEMM
// C[M x 512] = A[M x 512] (bf16, row-major) * Bt[512 x 512]^T (Bt is N-major bf16)
// One block = 8 waves = one 16-row M strip; wave w owns N tiles [4w, 4w+4).
// K loop is software-pipelined: fragments for step k+1 are loaded before the
// WMMAs of step k, so s_wait_loadcnt can retire a full K-step behind issue.
template <bool BIAS>
__global__ __launch_bounds__(256) void k_gemm_bf16(const bf16_t* __restrict__ A,
                                                   const bf16_t* __restrict__ Bt,
                                                   const float* __restrict__ bias,
                                                   float* __restrict__ Cout) {
    const int lane = threadIdx.x & 31;
    const int wave = threadIdx.x >> 5;
    const int m0   = blockIdx.x * 16;
    const int hf   = lane >> 4;    // K-half select (wave32 fragment layout)
    const int l16  = lane & 15;
    const int nt0  = wave * 4;     // first of 4 N tiles for this wave

    const bf16_t* arow = A + (size_t)(m0 + l16) * KK;
    const bf16_t* bp0  = Bt + (size_t)((nt0 + 0) * 16 + l16) * KK + hf * 16;
    const bf16_t* bp1  = Bt + (size_t)((nt0 + 1) * 16 + l16) * KK + hf * 16;
    const bf16_t* bp2  = Bt + (size_t)((nt0 + 2) * 16 + l16) * KK + hf * 16;
    const bf16_t* bp3  = Bt + (size_t)((nt0 + 3) * 16 + l16) * KK + hf * 16;

    v8f acc0 = {}, acc1 = {}, acc2 = {}, acc3 = {};

    // fragment double buffers
    V16U a[2], fb0[2], fb1[2], fb2[2], fb3[2];

    // prologue: load K-step 0 into buffer 0
    a[0].h[0]   = *(const v8bf*)(arow + hf * 16);
    a[0].h[1]   = *(const v8bf*)(arow + hf * 16 + 8);
    fb0[0].h[0] = *(const v8bf*)(bp0);     fb0[0].h[1] = *(const v8bf*)(bp0 + 8);
    fb1[0].h[0] = *(const v8bf*)(bp1);     fb1[0].h[1] = *(const v8bf*)(bp1 + 8);
    fb2[0].h[0] = *(const v8bf*)(bp2);     fb2[0].h[1] = *(const v8bf*)(bp2 + 8);
    fb3[0].h[0] = *(const v8bf*)(bp3);     fb3[0].h[1] = *(const v8bf*)(bp3 + 8);

#pragma unroll
    for (int s = 0; s < KK / 32; ++s) {
        const int cur = s & 1;
        const int nxt = cur ^ 1;
        if (s + 1 < KK / 32) {
            const int kk = (s + 1) * 32;
            __builtin_prefetch(arow + kk + 32, 0, 3);  // global_prefetch_b8 (next-next A slab)
            a[nxt].h[0]   = *(const v8bf*)(arow + kk + hf * 16);
            a[nxt].h[1]   = *(const v8bf*)(arow + kk + hf * 16 + 8);
            fb0[nxt].h[0] = *(const v8bf*)(bp0 + kk);  fb0[nxt].h[1] = *(const v8bf*)(bp0 + kk + 8);
            fb1[nxt].h[0] = *(const v8bf*)(bp1 + kk);  fb1[nxt].h[1] = *(const v8bf*)(bp1 + kk + 8);
            fb2[nxt].h[0] = *(const v8bf*)(bp2 + kk);  fb2[nxt].h[1] = *(const v8bf*)(bp2 + kk + 8);
            fb3[nxt].h[0] = *(const v8bf*)(bp3 + kk);  fb3[nxt].h[1] = *(const v8bf*)(bp3 + kk + 8);
        }
        acc0 = __builtin_amdgcn_wmma_f32_16x16x32_bf16(false, a[cur].v, false, fb0[cur].v, (short)0, acc0, false, false);
        acc1 = __builtin_amdgcn_wmma_f32_16x16x32_bf16(false, a[cur].v, false, fb1[cur].v, (short)0, acc1, false, false);
        acc2 = __builtin_amdgcn_wmma_f32_16x16x32_bf16(false, a[cur].v, false, fb2[cur].v, (short)0, acc2, false, false);
        acc3 = __builtin_amdgcn_wmma_f32_16x16x32_bf16(false, a[cur].v, false, fb3[cur].v, (short)0, acc3, false, false);
    }

    // C/D layout: VGPR r -> row r + 8*hf, col = l16 within tile
#pragma unroll
    for (int r = 0; r < 8; ++r) {
        float* crow = Cout + (size_t)(m0 + r + hf * 8) * CC;
        int c0 = (nt0 + 0) * 16 + l16, c1 = (nt0 + 1) * 16 + l16;
        int c2 = (nt0 + 2) * 16 + l16, c3 = (nt0 + 3) * 16 + l16;
        float v0 = acc0[r], v1 = acc1[r], v2 = acc2[r], v3 = acc3[r];
        if (BIAS) { v0 += bias[c0]; v1 += bias[c1]; v2 += bias[c2]; v3 += bias[c3]; }
        crow[c0] = v0; crow[c1] = v1; crow[c2] = v2; crow[c3] = v3;
    }
}

// ------------------------------------------------------------ aggregation
// agg[i][c] = dis[i]^2 * h_pre[i][c] + bias[c]   (self-loop message + bias)
__global__ __launch_bounds__(256) void k_agg_init(const float* __restrict__ dis,
                                                  const float* __restrict__ hpre,
                                                  const float* __restrict__ bias,
                                                  float* __restrict__ agg) {
    int idx = blockIdx.x * 256 + threadIdx.x;  // NN * 128 items
    if (idx >= NN * (CC / 4)) return;
    int i = idx >> 7;
    int c = (idx & 127) << 2;
    float d = dis[i], w = d * d;
    const float4 v = *(const float4*)(hpre + (size_t)i * CC + c);
    float4 o;
    o.x = w * v.x + bias[c + 0];
    o.y = w * v.y + bias[c + 1];
    o.z = w * v.z + bias[c + 2];
    o.w = w * v.w + bias[c + 3];
    *(float4*)(agg + (size_t)i * CC + c) = o;
}

// agg[dst][c] += norm[e] * h_pre[src][c]
__global__ __launch_bounds__(256) void k_scatter(const int* __restrict__ src,
                                                 const int* __restrict__ dst,
                                                 const float* __restrict__ nrm,
                                                 const float* __restrict__ hpre,
                                                 float* __restrict__ agg) {
    int idx = blockIdx.x * 256 + threadIdx.x;  // NE * 128 items
    if (idx >= NE * (CC / 4)) return;
    int e = idx >> 7;
    int c = (idx & 127) << 2;
    int s = src[e], d = dst[e];
    float w = nrm[e];
    const float4 v = *(const float4*)(hpre + (size_t)s * CC + c);
    float* ap = agg + (size_t)d * CC + c;
    atomicAdd(ap + 0, w * v.x);
    atomicAdd(ap + 1, w * v.y);
    atomicAdd(ap + 2, w * v.z);
    atomicAdd(ap + 3, w * v.w);
}

// ---------------------------------------------------------------- launcher
static inline size_t alignup(size_t x) { return (x + 255) & ~size_t(255); }

extern "C" void kernel_launch(void* const* d_in, const int* in_sizes, int n_in,
                              void* d_out, int out_size, void* d_ws, size_t ws_size,
                              hipStream_t stream) {
    const float* x     = (const float*)d_in[0];
    const int*   ei    = (const int*)d_in[1];
    const int*   src   = ei;
    const int*   dst   = ei + NE;
    const float* W1    = (const float*)d_in[2];
    const float* b1    = (const float*)d_in[3];
    const float* W2    = (const float*)d_in[4];
    const float* b2    = (const float*)d_in[5];
    const float* W3    = (const float*)d_in[6];
    const float* b3    = (const float*)d_in[7];
    const float* noise = (const float*)d_in[8];
    float* out = (float*)d_out;

    // workspace carve-up
    char* ws = (char*)d_ws;
    size_t off = 0;
    float*  dis   = (float*)(ws + off);  off += alignup(NN * 4);
    float*  nrm   = (float*)(ws + off);  off += alignup((size_t)NE * 4);
    bf16_t* Wt1   = (bf16_t*)(ws + off); off += alignup((size_t)KK * CC * 2);
    bf16_t* Wt2   = (bf16_t*)(ws + off); off += alignup((size_t)KK * CC * 2);
    bf16_t* Wt3   = (bf16_t*)(ws + off); off += alignup((size_t)KK * CC * 2);
    bf16_t* actbf = (bf16_t*)(ws + off); off += alignup((size_t)NN * CC * 2);  // x_bf16, reused as h_bf16
    bf16_t* nzbf  = (bf16_t*)(ws + off); off += alignup((size_t)MM * KK * 2);
    float*  hpre  = (float*)(ws + off);  off += alignup((size_t)NN * CC * 4);
    float*  agg   = (float*)(ws + off);  off += alignup((size_t)NN * CC * 4);

    const int NB   = 256;
    const int gN   = (NN + NB - 1) / NB;
    const int gE   = (NE + NB - 1) / NB;
    const int gW   = (KK * CC) / NB;           // 1024
    const int gXc  = (NN * CC + NB - 1) / NB;  // 20000
    const int gNz  = (MM * KK + NB - 1) / NB;  // 4000
    const int gAgg = (NN * (CC / 4)) / NB;     // 5000
    const int gSct = (NE * (CC / 4)) / NB;     // 80000
    const int gGmN = NN / 16;                  // 625 M-tiles
    const int gGmM = MM / 16;                  // 125 M-tiles

    // degree / normalization
    k_deg_init <<<gN, NB, 0, stream>>>(dis);
    k_deg_accum<<<gE, NB, 0, stream>>>(dst, dis);
    k_dis      <<<gN, NB, 0, stream>>>(dis);
    k_norm     <<<gE, NB, 0, stream>>>(src, dst, dis, nrm);

    // weight transpose+convert, activation convert
    k_w_transpose<<<gW, NB, 0, stream>>>(W1, Wt1);
    k_w_transpose<<<gW, NB, 0, stream>>>(W2, Wt2);
    k_w_transpose<<<gW, NB, 0, stream>>>(W3, Wt3);
    k_f32_to_bf16<<<gXc, NB, 0, stream>>>(x, actbf, NN * CC);
    k_f32_to_bf16<<<gNz, NB, 0, stream>>>(noise, nzbf, MM * KK);

    // layer 1: GEMM -> init(self loop + bias) -> scatter -> relu+bf16
    k_gemm_bf16<false><<<gGmN, NB, 0, stream>>>(actbf, Wt1, nullptr, hpre);
    k_agg_init<<<gAgg, NB, 0, stream>>>(dis, hpre, b1, agg);
    k_scatter <<<gSct, NB, 0, stream>>>(src, dst, nrm, hpre, agg);
    k_relu_to_bf16<<<gXc, NB, 0, stream>>>(agg, actbf, NN * CC);

    // layer 2: GEMM -> init -> scatter (into d_out) -> relu in place
    k_gemm_bf16<false><<<gGmN, NB, 0, stream>>>(actbf, Wt2, nullptr, hpre);
    k_agg_init<<<gAgg, NB, 0, stream>>>(dis, hpre, b2, out);
    k_scatter <<<gSct, NB, 0, stream>>>(src, dst, nrm, hpre, out);
    k_relu_inplace<<<gXc, NB, 0, stream>>>(out, NN * CC);

    // pred = noise @ W3 + b3 -> d_out rows [NN, NN+MM)
    k_gemm_bf16<true><<<gGmM, NB, 0, stream>>>(nzbf, Wt3, b3, out + (size_t)NN * CC);
}
